// LinearAttention_12111807775103
// MI455X (gfx1250) — matrix operand order repaired and checked by reading, not compiled, not run on hardware
//
#include <hip/hip_runtime.h>

// ---------------------------------------------------------------------------
// Types for CDNA5 WMMA (wave32): v_wmma_f32_16x16x32_bf16
// ---------------------------------------------------------------------------
typedef __attribute__((ext_vector_type(16))) __bf16 v16bf;
typedef __attribute__((ext_vector_type(8)))  __bf16 v8bf;
typedef __attribute__((ext_vector_type(8)))  float  v8f;

#define WMMA_BF16(a, b, c) \
    __builtin_amdgcn_wmma_f32_16x16x32_bf16(false, (a), false, (b), (short)0, (c), false, false)

// Problem constants
#define BATCH 4
#define SEQ   8192
#define DIM   512
#define HEADS 8
#define DH    64
#define MROWS (BATCH * SEQ)   // 32768

__device__ __forceinline__ unsigned short f2bf_u(float f) {
    unsigned u = __builtin_bit_cast(unsigned, f);
    u += 0x7FFFu + ((u >> 16) & 1u);           // round-to-nearest-even
    return (unsigned short)(u >> 16);
}

__device__ __forceinline__ v8f vzero8() {
    v8f z;
#pragma unroll
    for (int i = 0; i < 8; ++i) z[i] = 0.0f;
    return z;
}

// ---------------------------------------------------------------------------
// Kernel 0: pre-swizzle one weight matrix W[512,512] f32 into bf16 WMMA
// B-fragment order in global memory (L2-resident, reused by all row blocks):
//   Wsw[((ntile*16 + kstep)*32 + lane)*16 + j] = bf16(W[col, k])
//   with col = ntile*16 + (lane&15),  k = kstep*32 + 16*(lane>>4) + j.
// One thread produces one fragment lane: 16 consecutive f32 -> 16 bf16 (32B).
// ---------------------------------------------------------------------------
__global__ __launch_bounds__(256)
void wswz_kernel(const float* __restrict__ W, unsigned short* __restrict__ Wsw)
{
    const int t     = blockIdx.x * 256 + threadIdx.x;   // 0..16383
    const int lane  = t & 31;
    const int kstep = (t >> 5) & 15;
    const int ntile = t >> 9;                           // 0..31
    const int col   = ntile * 16 + (lane & 15);
    const int kbase = kstep * 32 + 16 * (lane >> 4);

    const float4* p = reinterpret_cast<const float4*>(W + (size_t)col * DIM + kbase);
    float4 f0 = p[0], f1 = p[1], f2 = p[2], f3 = p[3];
    float fa[16] = {f0.x, f0.y, f0.z, f0.w, f1.x, f1.y, f1.z, f1.w,
                    f2.x, f2.y, f2.z, f2.w, f3.x, f3.y, f3.z, f3.w};
    unsigned ua[8];
#pragma unroll
    for (int i = 0; i < 8; ++i)
        ua[i] = (unsigned)f2bf_u(fa[2 * i]) | ((unsigned)f2bf_u(fa[2 * i + 1]) << 16);

    uint4* q = reinterpret_cast<uint4*>(Wsw + (size_t)t * 16);
    q[0] = make_uint4(ua[0], ua[1], ua[2], ua[3]);
    q[1] = make_uint4(ua[4], ua[5], ua[6], ua[7]);
}

// ---------------------------------------------------------------------------
// Kernel 1: tiled GEMM  Y = act(X @ W^T + b)
//   X: [M, 512] (f32 or bf16)  Wsw: pre-swizzled bf16 B-fragments (global/L2)
//   Block: 256 thr (8 waves). Tile: 128 rows x 64 cols, K-step 32.
//   A tile staged in LDS pre-swizzled to the WMMA A fragment layout
//   (contiguous k, matching Wsw): row = lane&15, k = 16*(lane>>4) + j.
//   B fragments read straight from L2 (2x global_load_b128 per fragment).
// ---------------------------------------------------------------------------
template<bool ACT, bool IN_BF16, bool OUT_F32>
__global__ __launch_bounds__(256, 2)
void gemm_bias_kernel(const void* __restrict__ Xv,
                      const unsigned short* __restrict__ Wsw,
                      const float* __restrict__ bias, void* __restrict__ Yv)
{
    __shared__ __align__(32) unsigned short Asw[8][32][16];  // 8 KB

    const int tid  = threadIdx.x;
    const int lane = tid & 31;
    const int wave = tid >> 5;
    const int m0   = blockIdx.x * 128;
    const int n0   = blockIdx.y * 64;

    const float*          Xf = (const float*)Xv;
    const unsigned short* Xb = (const unsigned short*)Xv;

    v8f acc[4];
#pragma unroll
    for (int g = 0; g < 4; ++g) acc[g] = vzero8();

    const int arow = tid >> 1;            // 0..127
    const int akb  = (tid & 1) * 16;      // 0 / 16

    for (int kk = 0; kk < DIM; kk += 32) {
        // ---- stage X tile (128x32) -> A-swizzled bf16 LDS ----
        {
            unsigned short hv[16];
            if (IN_BF16) {
                const uint4* p = reinterpret_cast<const uint4*>(
                    Xb + (size_t)(m0 + arow) * DIM + kk + akb);
                uint4 u0 = p[0], u1 = p[1];
                unsigned ua[8] = {u0.x, u0.y, u0.z, u0.w, u1.x, u1.y, u1.z, u1.w};
#pragma unroll
                for (int i = 0; i < 8; ++i) {
                    hv[2 * i]     = (unsigned short)(ua[i] & 0xFFFFu);
                    hv[2 * i + 1] = (unsigned short)(ua[i] >> 16);
                }
                if (kk + 32 < DIM)
                    __builtin_prefetch(Xb + (size_t)(m0 + arow) * DIM + kk + 32 + akb, 0, 0);
            } else {
                const float4* p = reinterpret_cast<const float4*>(
                    Xf + (size_t)(m0 + arow) * DIM + kk + akb);
                float4 f0 = p[0], f1 = p[1], f2 = p[2], f3 = p[3];
                float fa[16] = {f0.x, f0.y, f0.z, f0.w, f1.x, f1.y, f1.z, f1.w,
                                f2.x, f2.y, f2.z, f2.w, f3.x, f3.y, f3.z, f3.w};
#pragma unroll
                for (int i = 0; i < 16; ++i) hv[i] = f2bf_u(fa[i]);
                if (kk + 32 < DIM)
                    __builtin_prefetch(Xf + (size_t)(m0 + arow) * DIM + kk + 32 + akb, 0, 0);
            }
#pragma unroll
            for (int i = 0; i < 16; ++i) {
                int k  = akb + i;
                int lh = k >> 4;
                int j  = k & 15;
                Asw[arow >> 4][(lh << 4) | (arow & 15)][j] = hv[i];
            }
        }
        __syncthreads();

        // ---- 4 WMMAs per wave per K-step; B straight from L2 ----
        v16bf a = *reinterpret_cast<const v16bf*>(&Asw[wave][lane][0]);
        const unsigned short* wbase =
            Wsw + ((size_t)(n0 >> 4) * 16 + (kk >> 5)) * 512 + (size_t)lane * 16;
#pragma unroll
        for (int g = 0; g < 4; ++g) {
            v16bf b = *reinterpret_cast<const v16bf*>(wbase + (size_t)g * 16 * 512);
            acc[g] = WMMA_BF16(a, b, acc[g]);
        }
        __syncthreads();
    }

    // ---- epilogue: bias (+ elu+1), scattered stores ----
    float*          Yf = (float*)Yv;
    unsigned short* Yb = (unsigned short*)Yv;
    const int rbase = m0 + wave * 16 + ((lane >> 4) << 3);
#pragma unroll
    for (int g = 0; g < 4; ++g) {
        const int col = n0 + g * 16 + (lane & 15);
        const float bvv = bias[col];
#pragma unroll
        for (int r = 0; r < 8; ++r) {
            float v = acc[g][r] + bvv;
            if (ACT) v = (v > 0.0f) ? (v + 1.0f) : __expf(v);  // elu(x)+1
            size_t off = (size_t)(rbase + r) * DIM + col;
            if (OUT_F32) Yf[off] = v;
            else         Yb[off] = f2bf_u(v);
        }
    }
}

// ---------------------------------------------------------------------------
// Kernel 2: context[b,h] += K_chunk^T @ V_chunk (WMMA), ksum from A-frags.
//   Grid: (16 T-chunks of 512, 32 bh).  Block: 128 thr (4 waves).
// ---------------------------------------------------------------------------
__global__ __launch_bounds__(128, 4)
void ctx_kernel(const unsigned short* __restrict__ kh,
                const unsigned short* __restrict__ vh,
                float* __restrict__ ctx, float* __restrict__ ksum)
{
    __shared__ __align__(32) unsigned short KA[4][32][16];  // A-swizzled K^T tile
    __shared__ __align__(32) unsigned short VB[4][32][16];  // B-swizzled V tile

    const int tid = threadIdx.x, lane = tid & 31, wave = tid >> 5;
    const int bh = blockIdx.y, b = bh >> 3, h = bh & 7;
    const int t0 = blockIdx.x * 512;
    const size_t rowbase = ((size_t)b * SEQ + t0) * DIM + h * DH;

    v8f acc[4];
#pragma unroll
    for (int g = 0; g < 4; ++g) acc[g] = vzero8();
    float ks_acc = 0.0f;

    const int tl = tid >> 2;          // local t 0..31
    const int cb = (tid & 3) * 16;    // col base 0/16/32/48

    for (int ts = 0; ts < 512; ts += 32) {
        const size_t gaddr = rowbase + (size_t)(ts + tl) * DIM + cb;
        // K tile -> A-swizzle (A row = channel c, A k = t)
        {
            const uint4* p = reinterpret_cast<const uint4*>(kh + gaddr);
            uint4 u0 = p[0], u1 = p[1];
            unsigned ua[8] = {u0.x, u0.y, u0.z, u0.w, u1.x, u1.y, u1.z, u1.w};
            const int lnb = (((tl >> 3) & 1) << 4);
            const int jb  = (tl & 7) + 8 * (tl >> 4);
#pragma unroll
            for (int i = 0; i < 8; ++i) {
                int c0 = cb + 2 * i, c1 = c0 + 1;
                KA[c0 >> 4][lnb | (c0 & 15)][jb] = (unsigned short)(ua[i] & 0xFFFFu);
                KA[c1 >> 4][lnb | (c1 & 15)][jb] = (unsigned short)(ua[i] >> 16);
            }
        }
        // V tile -> B-swizzle (B k = t, B col = d)
        {
            const uint4* p = reinterpret_cast<const uint4*>(vh + gaddr);
            uint4 u0 = p[0], u1 = p[1];
            unsigned ua[8] = {u0.x, u0.y, u0.z, u0.w, u1.x, u1.y, u1.z, u1.w};
            const int lnb = (((tl >> 3) & 1) << 4);
            const int jb  = (tl & 7) + 8 * (tl >> 4);
#pragma unroll
            for (int i = 0; i < 8; ++i) {
                int d0 = cb + 2 * i, d1 = d0 + 1;
                VB[d0 >> 4][lnb | (d0 & 15)][jb] = (unsigned short)(ua[i] & 0xFFFFu);
                VB[d1 >> 4][lnb | (d1 & 15)][jb] = (unsigned short)(ua[i] >> 16);
            }
        }
        __syncthreads();

        v16bf a = *reinterpret_cast<const v16bf*>(&KA[wave][lane][0]);
#pragma unroll
        for (int j = 0; j < 16; ++j) ks_acc += (float)a[j];   // free ksum partial
#pragma unroll
        for (int g = 0; g < 4; ++g) {
            v16bf bb = *reinterpret_cast<const v16bf*>(&VB[g][lane][0]);
            acc[g] = WMMA_BF16(a, bb, acc[g]);
        }
        __syncthreads();
    }

    float* cbase = ctx + (size_t)bh * DH * DH;
    const int rm = wave * 16 + ((lane >> 4) << 3);
#pragma unroll
    for (int g = 0; g < 4; ++g) {
        const int col = g * 16 + (lane & 15);
#pragma unroll
        for (int r = 0; r < 8; ++r)
            atomicAdd(&cbase[(size_t)(rm + r) * DH + col], acc[g][r]);
    }
    atomicAdd(&ksum[bh * DH + wave * 16 + (lane & 15)], ks_acc);
}

// ---------------------------------------------------------------------------
// Kernel 3: o = (Q @ context) / (Q . ksum)   per (b,h)
//   Grid: (64 row-blocks of 128, 32 bh).  Block: 128 thr (4 waves).
// ---------------------------------------------------------------------------
__global__ __launch_bounds__(128, 4)
void attn_out_kernel(const unsigned short* __restrict__ qh,
                     const float* __restrict__ ctx,
                     const float* __restrict__ ksum,
                     unsigned short* __restrict__ oh)
{
    __shared__ __align__(32) unsigned short CB[2][4][32][16];  // ctx, both K-steps
    __shared__ float KS[DH];
    __shared__ float DP[4][2][32];

    const int tid = threadIdx.x, lane = tid & 31, wave = tid >> 5;
    const int bh = blockIdx.y, b = bh >> 3, h = bh & 7;
    const int t0 = blockIdx.x * 128;
    const float* cb_g = ctx + (size_t)bh * DH * DH;

    // stage context (64x64 f32 -> bf16 B-swizzle) + ksum
    {
        const int c  = tid >> 1;
        const int db = (tid & 1) * 32;
        const int s  = c >> 5, kl = c & 31;
        const int ln = (((kl >> 3) & 1) << 4), jj = (kl & 7) + 8 * (kl >> 4);
#pragma unroll
        for (int i = 0; i < 32; i += 4) {
            float4 f = *reinterpret_cast<const float4*>(&cb_g[c * DH + db + i]);
            float fv[4] = {f.x, f.y, f.z, f.w};
#pragma unroll
            for (int q = 0; q < 4; ++q) {
                int d = db + i + q;
                CB[s][d >> 4][ln | (d & 15)][jj] = f2bf_u(fv[q]);
            }
        }
        if (tid < DH) KS[tid] = ksum[bh * DH + tid];
    }
    __syncthreads();

    const int lh = lane >> 4;
    const int m  = lane & 15;

    v8f acc[2][4];
#pragma unroll
    for (int i = 0; i < 2; ++i)
#pragma unroll
        for (int g = 0; g < 4; ++g) acc[i][g] = vzero8();

#pragma unroll
    for (int i = 0; i < 2; ++i) {
        const int row = t0 + wave * 32 + 16 * i + m;
        const unsigned short* qrow = qh + ((size_t)b * SEQ + row) * DIM + h * DH;
        float dacc = 0.0f;
#pragma unroll
        for (int s = 0; s < 2; ++s) {
            v8bf lo = *reinterpret_cast<const v8bf*>(qrow + 32 * s + 8 * lh);
            v8bf hi = *reinterpret_cast<const v8bf*>(qrow + 32 * s + 16 + 8 * lh);
            v16bf a;
#pragma unroll
            for (int j = 0; j < 8; ++j) { a[j] = lo[j]; a[8 + j] = hi[j]; }
#pragma unroll
            for (int j = 0; j < 8; ++j) {
                dacc += (float)lo[j] * KS[32 * s + 8 * lh + j];
                dacc += (float)hi[j] * KS[32 * s + 16 + 8 * lh + j];
            }
#pragma unroll
            for (int g = 0; g < 4; ++g) {
                v16bf bb = *reinterpret_cast<const v16bf*>(&CB[s][g][lane][0]);
                acc[i][g] = WMMA_BF16(a, bb, acc[i][g]);
            }
        }
        DP[wave][i][lane] = dacc;
    }
    __syncthreads();

#pragma unroll
    for (int i = 0; i < 2; ++i) {
#pragma unroll
        for (int g = 0; g < 4; ++g) {
            const int col = g * 16 + m;
#pragma unroll
            for (int r = 0; r < 8; ++r) {
                const int rowm  = r + 8 * lh;
                const float den = DP[wave][i][rowm] + DP[wave][i][rowm + 16];
                const float val = acc[i][g][r] / den;
                const int t = t0 + wave * 32 + 16 * i + rowm;
                oh[((size_t)b * SEQ + t) * DIM + h * DH + col] = f2bf_u(val);
            }
        }
    }
}

// ---------------------------------------------------------------------------
// Launch: weight pre-swizzle, memset accumulators, 3 projections, context,
// attention, output projection.
// ---------------------------------------------------------------------------
extern "C" void kernel_launch(void* const* d_in, const int* in_sizes, int n_in,
                              void* d_out, int out_size, void* d_ws, size_t ws_size,
                              hipStream_t stream)
{
    const float* q  = (const float*)d_in[0];
    const float* k  = (const float*)d_in[1];
    const float* v  = (const float*)d_in[2];
    const float* Wq = (const float*)d_in[3]; const float* bq = (const float*)d_in[4];
    const float* Wk = (const float*)d_in[5]; const float* bk = (const float*)d_in[6];
    const float* Wv = (const float*)d_in[7]; const float* bv = (const float*)d_in[8];
    const float* Wo = (const float*)d_in[9]; const float* bo = (const float*)d_in[10];

    char* ws = (char*)d_ws;
    const size_t elems   = (size_t)MROWS * DIM;          // 16M
    const size_t wbytes  = (size_t)DIM * DIM * 2;        // 512 KB per swizzled weight
    const size_t ctxpad  = ((size_t)1 << 20);            // 1 MB slot for ctx + ksum
    unsigned short* qh  = (unsigned short*)ws;           // bf16; later aliased as o
    unsigned short* kh  = (unsigned short*)(ws + 2 * elems);
    unsigned short* vh  = (unsigned short*)(ws + 4 * elems);
    float*          ctx = (float*)(ws + 6 * elems);      // [32][64][64]
    float*          ksm = (float*)(ws + 6 * elems + (size_t)32 * DH * DH * 4);
    unsigned short* wsq = (unsigned short*)(ws + 6 * elems + ctxpad);
    unsigned short* wsk = (unsigned short*)((char*)wsq + wbytes);
    unsigned short* wsv = (unsigned short*)((char*)wsq + 2 * wbytes);
    unsigned short* wso = (unsigned short*)((char*)wsq + 3 * wbytes);

    // pre-swizzle all four weight matrices into WMMA B-fragment order
    wswz_kernel<<<64, 256, 0, stream>>>(Wq, wsq);
    wswz_kernel<<<64, 256, 0, stream>>>(Wk, wsk);
    wswz_kernel<<<64, 256, 0, stream>>>(Wv, wsv);
    wswz_kernel<<<64, 256, 0, stream>>>(Wo, wso);

    (void)hipMemsetAsync(ctx, 0, ((size_t)32 * DH * DH + 32 * DH) * sizeof(float), stream);

    dim3 gg(MROWS / 128, DIM / 64);   // (256, 8)
    gemm_bias_kernel<true,  false, false><<<gg, 256, 0, stream>>>(q, wsq, bq, qh);
    gemm_bias_kernel<true,  false, false><<<gg, 256, 0, stream>>>(k, wsk, bk, kh);
    gemm_bias_kernel<false, false, false><<<gg, 256, 0, stream>>>(v, wsv, bv, vh);

    ctx_kernel<<<dim3(16, 32), 128, 0, stream>>>(kh, vh, ctx, ksm);

    attn_out_kernel<<<dim3(SEQ / 128, 32), 128, 0, stream>>>(qh, ctx, ksm, qh);

    gemm_bias_kernel<false, true, true><<<gg, 256, 0, stream>>>(qh, wso, bo, (float*)d_out);
}
